// HelixMemory_89928025243666
// MI455X (gfx1250) — matrix operand order repaired and checked by reading, not compiled, not run on hardware
//
#include <hip/hip_runtime.h>

typedef __attribute__((ext_vector_type(16))) __bf16 v16bf;
typedef __attribute__((ext_vector_type(8)))  float  v8f;
typedef __attribute__((ext_vector_type(4)))  float  f4;

namespace {
constexpr int kD       = 1024;   // feature dim
constexpr int kSeq     = 512;    // S
constexpr int kMemRows = 2558;   // MEM_LEN + N_CONV
constexpr int kChainM  = 254;    // batch-invariant compressed rows
constexpr int kKTot    = 2048;   // effective GEMM K (2 taps * D)
constexpr int TM = 128, TN = 128, TK = 32;
constexpr int LDS_STRIDE = TK + 8;   // 40 bf16 = 80B rows (16B aligned)
}

union Frag16 { v16bf v; uint4 q[2]; };
union Pack4  { __bf16 h[4]; uint2 u; };

__device__ __forceinline__ void split_bf16(float x, __bf16& h, __bf16& l) {
  h = (__bf16)x;               // high part (RNE)
  l = (__bf16)(x - (float)h);  // residual
}

// C = A(M x 2048) @ W(2048 x 1024), bf16x3 split-precision with fp32 accumulate.
// job 0..31: A = inputs[b] (M=256), dest = out[b] rows [254,510)
// job 32   : A = memory+2*D (M=254), dest = chain workspace
__global__ __launch_bounds__(256) void helix_gemm(
    const float* __restrict__ inputs,
    const float* __restrict__ memory,
    const float* __restrict__ filters,
    float* __restrict__ out,
    float* __restrict__ chain)
{
  __shared__ __align__(16) __bf16 Ahi[TM][LDS_STRIDE];
  __shared__ __align__(16) __bf16 Alo[TM][LDS_STRIDE];
  __shared__ __align__(16) __bf16 Bhi[TN][LDS_STRIDE];   // stored transposed: [n][k]
  __shared__ __align__(16) __bf16 Blo[TN][LDS_STRIDE];

  const int tid  = threadIdx.x;
  const int lane = tid & 31;
  const int wave = tid >> 5;      // 8 waves
  const int wm   = wave >> 2;     // 0..1 -> 64 rows each
  const int wn   = wave & 3;      // 0..3 -> 32 cols each

  const int job    = blockIdx.z;        // 0..32
  const int tile_m = blockIdx.y * TM;   // 0,128
  const int tile_n = blockIdx.x * TN;   // 0..896

  const float* Asrc;
  float*       Dst;
  int          Mrows;
  bool         ntC;
  if (job < 32) {
    Asrc  = inputs + (size_t)job * kSeq * kD;
    Dst   = out + (size_t)job * kMemRows * kD + (size_t)254 * kD;
    Mrows = 256;
    ntC   = true;    // streamed output, don't pollute caches
  } else {
    Asrc  = memory + 2 * kD;   // rows 2..509 of memory, viewed as (254 x 2048)
    Dst   = chain;
    Mrows = kChainM;
    ntC   = false;   // re-read 32x by assemble kernel, keep resident
  }

  // staging roles: A tile 128x32 (2 threads/row, 16 floats each)
  const int am = tid >> 1;
  const int ak = (tid & 1) * 16;
  // W tile 32x128: 4(k) x 4(n) sub-block per thread -> packed b64 transposed stores
  const int bkg = (tid >> 5) * 4;   // k0: 0,4,...,28
  const int bng = (tid & 31) * 4;   // n0: 0,4,...,124

  const float* aptr = Asrc + (size_t)(tile_m + am) * kKTot + ak;
  const float* wptr = filters + (size_t)bkg * kD + tile_n + bng;

  v8f acc[4][2] = {};

  for (int ks = 0; ks < kKTot / TK; ++ks) {
    // ---- stage A (fp32 -> bf16 hi/lo), contiguous stores (merge to b128) ----
    const float* ap = aptr + ks * TK;
    #pragma unroll
    for (int v = 0; v < 4; ++v) {
      float4 x = *(const float4*)(ap + v * 4);
      __bf16 h, l;
      split_bf16(x.x, h, l); Ahi[am][ak + v*4 + 0] = h; Alo[am][ak + v*4 + 0] = l;
      split_bf16(x.y, h, l); Ahi[am][ak + v*4 + 1] = h; Alo[am][ak + v*4 + 1] = l;
      split_bf16(x.z, h, l); Ahi[am][ak + v*4 + 2] = h; Alo[am][ak + v*4 + 2] = l;
      split_bf16(x.w, h, l); Ahi[am][ak + v*4 + 3] = h; Alo[am][ak + v*4 + 3] = l;
    }
    // ---- stage W transposed (fp32 -> bf16 hi/lo), packed b64 stores ----
    const float* wp = wptr + (size_t)ks * TK * kD;
    {
      Pack4 ph[4], pl[4];            // [j] = n offset, .h[i] = k offset
      #pragma unroll
      for (int i = 0; i < 4; ++i) {
        float4 x = *(const float4*)(wp + (size_t)i * kD);
        __bf16 h, l;
        split_bf16(x.x, h, l); ph[0].h[i] = h; pl[0].h[i] = l;
        split_bf16(x.y, h, l); ph[1].h[i] = h; pl[1].h[i] = l;
        split_bf16(x.z, h, l); ph[2].h[i] = h; pl[2].h[i] = l;
        split_bf16(x.w, h, l); ph[3].h[i] = h; pl[3].h[i] = l;
      }
      #pragma unroll
      for (int j = 0; j < 4; ++j) {
        *(uint2*)&Bhi[bng + j][bkg] = ph[j].u;
        *(uint2*)&Blo[bng + j][bkg] = pl[j].u;
      }
    }
    if (ks + 1 < kKTot / TK) {
      __builtin_prefetch(ap + TK, 0, 3);                  // next A k-slab
      __builtin_prefetch(wp + (size_t)TK * kD, 0, 3);     // next W k-slab
    }
    __syncthreads();

    // ---- load fragments per ISA VGPR layouts ----
    const int mrow = lane & 15;
    const int kA   = (lane >> 4) << 3;   // A: lanes 16-31 hold K+8
    const int kB   = (lane >> 4) << 3;   // B: lanes 16-31 hold K+8

    Frag16 fah[4], fal[4];
    #pragma unroll
    for (int sm = 0; sm < 4; ++sm) {
      const int r = wm * 64 + sm * 16 + mrow;
      fah[sm].q[0] = *(const uint4*)&Ahi[r][kA];
      fah[sm].q[1] = *(const uint4*)&Ahi[r][kA + 16];
      fal[sm].q[0] = *(const uint4*)&Alo[r][kA];
      fal[sm].q[1] = *(const uint4*)&Alo[r][kA + 16];
    }
    Frag16 fbh[2], fbl[2];
    #pragma unroll
    for (int sn = 0; sn < 2; ++sn) {
      const int c = wn * 32 + sn * 16 + mrow;
      fbh[sn].q[0] = *(const uint4*)&Bhi[c][kB];
      fbh[sn].q[1] = *(const uint4*)&Bhi[c][kB + 16];
      fbl[sn].q[0] = *(const uint4*)&Blo[c][kB];
      fbl[sn].q[1] = *(const uint4*)&Blo[c][kB + 16];
    }

    // ---- bf16x3: hi*hi + hi*lo + lo*hi, fp32 accumulate ----
    #pragma unroll
    for (int sm = 0; sm < 4; ++sm) {
      #pragma unroll
      for (int sn = 0; sn < 2; ++sn) {
        acc[sm][sn] = __builtin_amdgcn_wmma_f32_16x16x32_bf16(
            false, fah[sm].v, false, fbh[sn].v, (short)0, acc[sm][sn], false, false);
        acc[sm][sn] = __builtin_amdgcn_wmma_f32_16x16x32_bf16(
            false, fah[sm].v, false, fbl[sn].v, (short)0, acc[sm][sn], false, false);
        acc[sm][sn] = __builtin_amdgcn_wmma_f32_16x16x32_bf16(
            false, fal[sm].v, false, fbh[sn].v, (short)0, acc[sm][sn], false, false);
      }
    }
    __syncthreads();
  }

  // ---- store C per 16x16 f32 layout: VGPR r -> M = r + 8*(lane>=16), N = lane&15 ----
  const int ccol = lane & 15;
  const int roff = (lane >> 4) << 3;
  #pragma unroll
  for (int sm = 0; sm < 4; ++sm) {
    #pragma unroll
    for (int sn = 0; sn < 2; ++sn) {
      const int gc = tile_n + wn * 32 + sn * 16 + ccol;
      #pragma unroll
      for (int r = 0; r < 8; ++r) {
        const int gr = tile_m + wm * 64 + sm * 16 + roff + r;
        if (gr < Mrows) {
          float* p = &Dst[(size_t)gr * kD + gc];
          if (ntC) __builtin_nontemporal_store(acc[sm][sn][r], p);
          else     *p = acc[sm][sn][r];
        }
      }
    }
  }
}

// Assemble the copy portions of new_mem for every batch:
//   rows [0,254)      <- chain workspace (batch-invariant compress results)
//   rows [510,2046)   <- memory[1022:2558]
//   rows [2046,2558)  <- inputs[b]
__global__ __launch_bounds__(256) void helix_assemble(
    const f4* __restrict__ inputs,
    const f4* __restrict__ memory,
    const f4* __restrict__ chain,
    f4* __restrict__ out)
{
  const unsigned PER_B = 2302u * 256u;            // rows copied per batch * float4/row
  unsigned gid = blockIdx.x * 256u + threadIdx.x; // total = 32*PER_B exactly
  unsigned b   = gid / PER_B;
  unsigned rem = gid - b * PER_B;
  unsigned r   = rem >> 8;     // 256 float4 per 1024-float row
  unsigned c   = rem & 255u;

  f4 v;
  unsigned drow;
  if (r < 254u) {                    // chain: small, reused 32x -> regular load
    v    = chain[(size_t)r * 256u + c];
    drow = r;
  } else if (r < 1790u) {            // memory window: reused 32x -> regular load
    v    = memory[(size_t)(1022u + (r - 254u)) * 256u + c];
    drow = 510u + (r - 254u);
  } else {                           // inputs: read-once -> non-temporal
    v    = __builtin_nontemporal_load(
              &inputs[((size_t)b * 512u + (r - 1790u)) * 256u + c]);
    drow = 2046u + (r - 1790u);
  }
  // streamed 335MB output -> non-temporal store
  __builtin_nontemporal_store(v, &out[((size_t)b * 2558u + drow) * 256u + c]);
}

extern "C" void kernel_launch(void* const* d_in, const int* in_sizes, int n_in,
                              void* d_out, int out_size, void* d_ws, size_t ws_size,
                              hipStream_t stream) {
  const float* inputs  = (const float*)d_in[0];   // (32, 512, 1024) fp32
  const float* memory  = (const float*)d_in[1];   // (2558, 1024) fp32
  const float* filters = (const float*)d_in[2];   // (2, 1024, 1024) fp32 == W (2048,1024)
  float* out   = (float*)d_out;                   // (32, 2558, 1024) fp32
  float* chain = (float*)d_ws;                    // 254*1024 floats scratch

  // 33 jobs x 2 M-tiles x 8 N-tiles
  dim3 grid(kD / TN, 256 / TM, 33);
  helix_gemm<<<grid, 256, 0, stream>>>(inputs, memory, filters, out, chain);

  const unsigned totalVec4 = 32u * 2302u * 256u;  // divisible by 256
  helix_assemble<<<dim3(totalVec4 / 256u), 256, 0, stream>>>(
      (const f4*)inputs, (const f4*)memory, (const f4*)chain, (f4*)out);
}
